// NeuralOT_15083925143897
// MI455X (gfx1250) — matrix operand order repaired and checked by reading, not compiled, not run on hardware
//
#include <hip/hip_runtime.h>
#include <hip/hip_bf16.h>

typedef __attribute__((ext_vector_type(16))) __bf16 v16bf;
typedef __attribute__((ext_vector_type(8)))  __bf16 v8bf;
typedef __attribute__((ext_vector_type(8)))  float  v8f;

#define EPS  0.1f
#define NN   1024
#define DD   12288          // 3*64*64
#define BK   32
#define KT   (DD / BK)      // 384 K-steps
#define LDA  40             // padded LDS row stride in bf16 elems (80 B) -> conflict-free b128 lane pattern

// workspace layout (floats)
#define WS_ACC 0
#define WS_X2  16
#define WS_Y2  (16 + NN)
#define WS_U   (16 + 2 * NN)
#define WS_V   (16 + 3 * NN)

__device__ __forceinline__ void gload4(const float* __restrict__ p, float4 r[4]) {
    const float4* q = reinterpret_cast<const float4*>(p);   // global_load_b128 x4
    r[0] = q[0]; r[1] = q[1]; r[2] = q[2]; r[3] = q[3];
}

__device__ __forceinline__ void cvt_store16(__bf16* dst, const float4 r[4]) {
    v8bf a, b;
    a[0] = (__bf16)r[0].x; a[1] = (__bf16)r[0].y; a[2] = (__bf16)r[0].z; a[3] = (__bf16)r[0].w;
    a[4] = (__bf16)r[1].x; a[5] = (__bf16)r[1].y; a[6] = (__bf16)r[1].z; a[7] = (__bf16)r[1].w;
    b[0] = (__bf16)r[2].x; b[1] = (__bf16)r[2].y; b[2] = (__bf16)r[2].z; b[3] = (__bf16)r[2].w;
    b[4] = (__bf16)r[3].x; b[5] = (__bf16)r[3].y; b[6] = (__bf16)r[3].z; b[7] = (__bf16)r[3].w;
    *reinterpret_cast<v8bf*>(dst)     = a;                  // ds_store_b128
    *reinterpret_cast<v8bf*>(dst + 8) = b;                  // ds_store_b128
}

// ---------------------------------------------------------------------------
// Kernel 1: row norms + table gathers + accumulator reset
// blocks 0..1023 handle x rows, 1024..2047 handle y rows
// ---------------------------------------------------------------------------
__global__ __launch_bounds__(256)
void ot_prep(const int* __restrict__ x_idx, const float* __restrict__ x,
             const int* __restrict__ y_idx, const float* __restrict__ y,
             const float* __restrict__ ut, const float* __restrict__ vt,
             float* __restrict__ ws)
{
    __shared__ float red[256];
    const int b = blockIdx.x;
    const int t = threadIdx.x;
    const bool isX = (b < NN);
    const int row = isX ? b : (b - NN);
    const float* p = (isX ? x : y) + (size_t)row * DD;

    float s = 0.f;
    for (int i = t; i < DD; i += 256) { float v = p[i]; s += v * v; }
    red[t] = s;
    __syncthreads();
    for (int off = 128; off > 0; off >>= 1) {
        if (t < off) red[t] += red[t + off];
        __syncthreads();
    }
    if (t == 0) {
        if (isX) {
            ws[WS_X2 + row] = red[0];
            ws[WS_U  + row] = ut[x_idx[row]];
        } else {
            ws[WS_Y2 + row] = red[0];
            ws[WS_V  + row] = vt[y_idx[row]];
        }
        if (b == 0) ws[WS_ACC] = 0.f;   // only block 0 touches the accumulator here
    }
}

// ---------------------------------------------------------------------------
// Kernel 2: bf16 WMMA GEMM (xy^T) + fused entropic-OT epilogue + reduction
// grid (8,8), 256 threads = 8 waves of 32; block tile 128x128, wave tile 32x64
// ---------------------------------------------------------------------------
__global__ __launch_bounds__(256)
void ot_gemm(const float* __restrict__ x, const float* __restrict__ y,
             float* __restrict__ ws)
{
    __shared__ __bf16 As[2][128 * LDA];   // 128 rows of x tile, bf16, padded stride
    __shared__ __bf16 Bs[2][128 * LDA];   // 128 rows of y tile
    __shared__ float  red[256];

    const int tid  = threadIdx.x;
    const int lane = tid & 31;
    const int wid  = tid >> 5;
    const int wm   = wid >> 1;            // 0..3 : wave row (32 rows each)
    const int wn   = wid & 1;             // 0..1 : wave col (64 cols each)
    const int bm   = blockIdx.x;          // 0..7
    const int bn   = blockIdx.y;          // 0..7

    // staging assignment: thread -> (row 0..127, half 0/1 of the 32-float K slab)
    const int srow  = tid >> 1;
    const int shalf = tid & 1;
    const size_t xbase = (size_t)(bm * 128 + srow) * DD + (size_t)shalf * 16;
    const size_t ybase = (size_t)(bn * 128 + srow) * DD + (size_t)shalf * 16;
    const int sdst = srow * LDA + shalf * 16;

    const int laneRow = lane & 15;
    const int laneHi  = lane >> 4;        // 0: lanes 0-15, 1: lanes 16-31

    v8f acc[2][4];
    #pragma unroll
    for (int mi = 0; mi < 2; ++mi)
        #pragma unroll
        for (int ni = 0; ni < 4; ++ni)
            #pragma unroll
            for (int r = 0; r < 8; ++r)
                acc[mi][ni][r] = 0.f;

    // ---- prologue: stage K-step 0 into buffer 0 ----
    {
        float4 xr[4], yr[4];
        gload4(x + xbase, xr);
        gload4(y + ybase, yr);
        cvt_store16(&As[0][sdst], xr);
        cvt_store16(&Bs[0][sdst], yr);
    }
    __syncthreads();

    // ---- main K loop, double-buffered ----
    for (int kt = 0; kt < KT; ++kt) {
        const int p = kt & 1;
        const bool hasNext = (kt + 1 < KT);
        float4 xr[4], yr[4];
        if (hasNext) {                                     // prefetch next slab into regs
            gload4(x + xbase + (size_t)(kt + 1) * BK, xr);
            gload4(y + ybase + (size_t)(kt + 1) * BK, yr);
        }

        // A fragments: 16x32 bf16; lanes 0-15: K0..7 & K16..23, lanes 16-31: K8..15 & K24..31
        v16bf afrag[2];
        #pragma unroll
        for (int mi = 0; mi < 2; ++mi) {
            const int row = wm * 32 + mi * 16 + laneRow;
            const __bf16* ap = &As[p][row * LDA + laneHi * 8];
            v8bf lo = *reinterpret_cast<const v8bf*>(ap);        // ds_load_b128
            v8bf hi = *reinterpret_cast<const v8bf*>(ap + 16);   // ds_load_b128
            afrag[mi] = __builtin_shufflevector(lo, hi,
                0,1,2,3,4,5,6,7,8,9,10,11,12,13,14,15);
        }
        // B fragments: 32x16 bf16; lanes 0-15: K0..7 & K16..23, lanes 16-31: K8..15 & K24..31 (col = y row)
        #pragma unroll
        for (int ni = 0; ni < 4; ++ni) {
            const int col = wn * 64 + ni * 16 + laneRow;
            const __bf16* bp = &Bs[p][col * LDA + laneHi * 8];
            v8bf lo = *reinterpret_cast<const v8bf*>(bp);        // ds_load_b128
            v8bf hi = *reinterpret_cast<const v8bf*>(bp + 16);   // ds_load_b128
            v16bf bfrag = __builtin_shufflevector(lo, hi,
                0,1,2,3,4,5,6,7,8,9,10,11,12,13,14,15);
            #pragma unroll
            for (int mi = 0; mi < 2; ++mi) {
                acc[mi][ni] = __builtin_amdgcn_wmma_f32_16x16x32_bf16(
                    false, afrag[mi], false, bfrag,
                    (short)0, acc[mi][ni], false, false);
            }
        }

        if (hasNext) {                                     // fill the other buffer
            cvt_store16(&As[p ^ 1][sdst], xr);
            cvt_store16(&Bs[p ^ 1][sdst], yr);
        }
        __syncthreads();
    }

    // ---- fused epilogue: c = x2 + y2 - 2*dot ; term = m - eps*exp((m-c)/eps) ----
    const float* x2 = ws + WS_X2;
    const float* y2 = ws + WS_Y2;
    const float* uu = ws + WS_U;
    const float* vv = ws + WS_V;

    float local = 0.f;
    #pragma unroll
    for (int mi = 0; mi < 2; ++mi) {
        const int rowBase = bm * 128 + wm * 32 + mi * 16 + laneHi * 8;
        #pragma unroll
        for (int ni = 0; ni < 4; ++ni) {
            const int col  = bn * 128 + wn * 64 + ni * 16 + laneRow;
            const float y2c = y2[col];
            const float vc  = vv[col];
            #pragma unroll
            for (int r = 0; r < 8; ++r) {
                const int row = rowBase + r;              // C/D layout: VGPR r -> M = r (+8 hi half)
                const float c = x2[row] + y2c - 2.0f * acc[mi][ni][r];
                const float m = uu[row] + vc;
                local += m - EPS * __expf((m - c) * (1.0f / EPS));
            }
        }
    }

    red[tid] = local;
    __syncthreads();
    for (int off = 128; off > 0; off >>= 1) {
        if (tid < off) red[tid] += red[tid + off];
        __syncthreads();
    }
    if (tid == 0) atomicAdd(&ws[WS_ACC], red[0]);
}

// ---------------------------------------------------------------------------
// Kernel 3: scalar finalize: out = -mean
// ---------------------------------------------------------------------------
__global__ void ot_finalize(const float* __restrict__ ws, float* __restrict__ out)
{
    if (threadIdx.x == 0 && blockIdx.x == 0)
        out[0] = -ws[WS_ACC] * (1.0f / ((float)NN * (float)NN));
}

extern "C" void kernel_launch(void* const* d_in, const int* in_sizes, int n_in,
                              void* d_out, int out_size, void* d_ws, size_t ws_size,
                              hipStream_t stream)
{
    const int*   x_idx = (const int*)  d_in[0];
    const float* x     = (const float*)d_in[1];
    const int*   y_idx = (const int*)  d_in[2];
    const float* y     = (const float*)d_in[3];
    const float* ut    = (const float*)d_in[4];
    const float* vt    = (const float*)d_in[5];
    float* ws  = (float*)d_ws;
    float* out = (float*)d_out;

    ot_prep<<<2 * NN, 256, 0, stream>>>(x_idx, x, y_idx, y, ut, vt, ws);
    dim3 grid(8, 8);
    ot_gemm<<<grid, 256, 0, stream>>>(x, y, ws);
    ot_finalize<<<1, 64, 0, stream>>>(ws, out);
}
